// HA_15891378995287
// MI455X (gfx1250) — compile-verified, matrix operand order checked by reading, not run on hardware
//
#include <hip/hip_runtime.h>
#include <math.h>

// ---------------------------------------------------------------------------
// HA soft-attention mask for MI455X (gfx1250, wave32, WMMA).
//
// 31x31 gaussian = outer(v,v) -> exactly separable; two 1-D 31-tap passes.
// Each 16x16 tile of a 1-D pass is a GEMM vs a Toeplitz weight matrix:
// two v_wmma_f32_16x16x32_bf16 per tile, window based at out0-16 so each
// CDNA5 A-fragment lane is three disjoint 8-element 16B-aligned runs
// [x0-16,x0) [x0,x0+16) [x0+16,x0+32); runs are entirely in- or out-of-
// bounds -> border handling is a SCALAR branch substituting a zero run.
// The Toeplitz weight fragments are a pure function of the lane id, so they
// are built ONCE in the prep kernel (2KB table) and fetched with four 16B
// loads per conv wave. Pass 1 writes the intermediate transposed, bf16,
// with zeroed 16-elem halos (pitch 544) so pass 2 has no bounds logic.
// One tile per wave, exact launches, no loops, no per-element selects.
// ---------------------------------------------------------------------------

typedef __attribute__((ext_vector_type(16))) __bf16 v16bf;
typedef __attribute__((ext_vector_type(8)))  __bf16 v8bf;
typedef __attribute__((ext_vector_type(8)))  float  v8f;

#define IMG     64
#define DIM     512
#define TPITCH  544            // 16 halo + 512 + 16 halo (bf16 elems)
#define EPS_    1e-3f

__device__ __forceinline__ int f2key(float f) {
  int b = __float_as_int(f);
  return (b >= 0) ? b : (b ^ 0x7fffffff);
}
__device__ __forceinline__ float key2f(int k) {
  int b = (k >= 0) ? k : (k ^ 0x7fffffff);
  return __int_as_float(b);
}
__device__ __forceinline__ float clip01(float f) {
  return fminf(fmaxf(f, 0.0f), 1.0f);
}
// CDNA5 16-bit fragment K enumeration (ISA 7.12.2): lane-half h holds
// elems [0..7] = K 8h..8h+7, elems [8..15] = K 16+8h..16+8h+7.
__device__ __forceinline__ int kidx(int i, int h) {
  return ((i < 4) ? (2 * i) : (16 + 2 * (i - 4))) + 8 * h;
}

__device__ __forceinline__ v8bf cvt8c(float4 a, float4 b) {
  v8bf o;
  o[0] = (__bf16)clip01(a.x); o[1] = (__bf16)clip01(a.y);
  o[2] = (__bf16)clip01(a.z); o[3] = (__bf16)clip01(a.w);
  o[4] = (__bf16)clip01(b.x); o[5] = (__bf16)clip01(b.y);
  o[6] = (__bf16)clip01(b.z); o[7] = (__bf16)clip01(b.w);
  return o;
}
__device__ __forceinline__ v16bf cat16(v8bf lo, v8bf hi) {
  return __builtin_shufflevector(lo, hi, 0, 1, 2, 3, 4, 5, 6, 7,
                                 8, 9, 10, 11, 12, 13, 14, 15);
}
__device__ __forceinline__ v8bf zero8() {
  v8bf z;
#pragma unroll
  for (int e = 0; e < 8; ++e) z[e] = (__bf16)0.0f;
  return z;
}
// fetch the per-lane precomputed weight fragments (4 x 16B loads)
__device__ __forceinline__ void load_wfrag(const __bf16* __restrict__ wf,
                                           int lane, v16bf& b0, v16bf& b1) {
  const v8bf* p0 = (const v8bf*)wf + lane * 2;        // chunk 0 table
  const v8bf* p1 = (const v8bf*)(wf + 512) + lane * 2;  // chunk 1 table
  b0 = cat16(p0[0], p0[1]);
  b1 = cat16(p1[0], p1[1]);
}

// ---------------------------------------------------------------------------
// Kernel 0a: 1-D weights w[i] = sqrt(K2d[i][i]) (sum==1, since the 2-D kernel
// is outer(w,w)); build per-lane Toeplitz weight fragments; reset minmax keys.
// Window base = out0-16: weight index wi = 32*chunk + K - j - 1.
// ---------------------------------------------------------------------------
__global__ void ha_prep(const float* __restrict__ gk, __bf16* __restrict__ wf,
                        int* __restrict__ mm) {
  __shared__ float sw[31];
  const int t = threadIdx.x;
  if (t < 31) sw[t] = sqrtf(gk[t * 31 + t]);
  if (t < IMG) {
    mm[2 * t + 0] = 0x7fffffff;
    mm[2 * t + 1] = 0x80000000;
  }
  __syncthreads();
  if (t < 32) {                       // t == lane id
    const int h = (t >> 4) & 1;
    const int r = t & 15;
#pragma unroll
    for (int i = 0; i < 8; ++i) {
      const int Kb = kidx(i, h);
#pragma unroll
      for (int p = 0; p < 2; ++p) {
        const int K   = Kb + p;
        const int wi0 = K - r - 1;
        const int wi1 = 32 + K - r - 1;
        wf[t * 16 + 2 * i + p] =
            (__bf16)((wi0 >= 0 && wi0 <= 30) ? sw[wi0] : 0.0f);
        wf[512 + t * 16 + 2 * i + p] =
            (__bf16)((wi1 >= 0 && wi1 <= 30) ? sw[wi1] : 0.0f);
      }
    }
  }
}

// ---------------------------------------------------------------------------
// Kernel 0b: zero the 16-elem halos of every tmpT row (2MB, once per launch).
// ---------------------------------------------------------------------------
__global__ void __launch_bounds__(256)
ha_zhalo(__bf16* __restrict__ tmpT) {
  const int row = blockIdx.x * blockDim.x + threadIdx.x;  // 64*512 rows
  __bf16* p = tmpT + (size_t)row * TPITCH;
  const v8bf z = zero8();
  ((v8bf*)p)[0] = z;
  ((v8bf*)p)[1] = z;
  ((v8bf*)(p + 528))[0] = z;
  ((v8bf*)(p + 528))[1] = z;
}

// ---------------------------------------------------------------------------
// Kernel 1: horizontal pass, f32 input -> bf16 transposed+haloed tmpT.
// tmpT[x][16+y] = sum_k w[k]*clip(att[y][x+k-15]).
// Block b: img = b>>7, tx = (b>>2)&31 (SCALAR), ty = (b&3)*8 + wave.
// ---------------------------------------------------------------------------
__global__ void __launch_bounds__(256)
ha_hconv(const float* __restrict__ att, const __bf16* __restrict__ wf,
         __bf16* __restrict__ tmpT) {
  const int lane = threadIdx.x & 31;
  const int h    = (lane >> 4) & 1;
  const int r    = lane & 15;

  v16bf b0, b1;
  load_wfrag(wf, lane, b0, b1);
  const v8bf z8 = zero8();

  const int b   = blockIdx.x;
  const int tx  = (b >> 2) & 31;                       // scalar per block
  const int img = b >> 7;
  const int ty  = (b & 3) * 8 + (threadIdx.x >> 5);
  const int x0 = tx * 16, y0 = ty * 16;

  const float* __restrict__ row =
      att + (size_t)img * DIM * DIM + (size_t)(y0 + r) * DIM;

  // three 8-elem runs: [x0-16,x0) [x0,x0+16) [x0+16,x0+32), lane-half off 8h
  v8bf f0 = z8, f2 = z8;
  if (tx > 0) {                                        // scalar branch
    const float4* p = (const float4*)(row + (x0 - 16 + 8 * h));
    f0 = cvt8c(p[0], p[1]);
  }
  const float4* p = (const float4*)(row + (x0 + 8 * h));
  const v8bf f1 = cvt8c(p[0], p[1]);
  if (tx < 31) {                                       // scalar branch
    const float4* q = (const float4*)(row + (x0 + 16 + 8 * h));
    f2 = cvt8c(q[0], q[1]);
  }
  const v16bf a0 = cat16(f0, f1);
  const v16bf a1 = cat16(f2, z8);

  v8f acc = {};
  acc = __builtin_amdgcn_wmma_f32_16x16x32_bf16(false, a0, false, b0,
                                                (short)0, acc, false, false);
  acc = __builtin_amdgcn_wmma_f32_16x16x32_bf16(false, a1, false, b1,
                                                (short)0, acc, false, false);

  // D: lane (h,r), VGPR g -> y = y0+8h+g, x = x0+r. Transposed bf16 store.
  v8bf o;
#pragma unroll
  for (int g = 0; g < 8; ++g) o[g] = (__bf16)acc[g];
  *(v8bf*)(tmpT + (size_t)img * DIM * TPITCH + (size_t)(x0 + r) * TPITCH +
           (16 + y0 + 8 * h)) = o;
}

// ---------------------------------------------------------------------------
// Kernel 2: vertical pass = horizontal conv over tmpT (bf16, haloed) -> soft
// (f32) + fused per-image min/max. No bounds logic at all.
// ---------------------------------------------------------------------------
__global__ void __launch_bounds__(256)
ha_vconv(const __bf16* __restrict__ tmpT, const __bf16* __restrict__ wf,
         float* __restrict__ soft, int* __restrict__ mm) {
  const int lane = threadIdx.x & 31;
  const int h    = (lane >> 4) & 1;
  const int r    = lane & 15;

  v16bf b0, b1;
  load_wfrag(wf, lane, b0, b1);
  const v8bf z8 = zero8();

  const int b   = blockIdx.x;
  const int tX  = (b >> 2) & 31;
  const int img = b >> 7;
  const int tY  = (b & 3) * 8 + (threadIdx.x >> 5);
  const int X0 = tX * 16, Y0 = tY * 16;

  // haloed row: index 16+(Y0-16+8h+...) = Y0+8h+...  (always in [0,544))
  const __bf16* __restrict__ row =
      tmpT + (size_t)img * DIM * TPITCH + (size_t)(X0 + r) * TPITCH;

  const v8bf u0 = *(const v8bf*)(row + (Y0 + 8 * h));
  const v8bf u1 = *(const v8bf*)(row + (Y0 + 16 + 8 * h));
  const v8bf u2 = *(const v8bf*)(row + (Y0 + 32 + 8 * h));
  const v16bf a0 = cat16(u0, u1);
  const v16bf a1 = cat16(u2, z8);

  v8f acc = {};
  acc = __builtin_amdgcn_wmma_f32_16x16x32_bf16(false, a0, false, b0,
                                                (short)0, acc, false, false);
  acc = __builtin_amdgcn_wmma_f32_16x16x32_bf16(false, a1, false, b1,
                                                (short)0, acc, false, false);

  // D: lane (h,r), VGPR g -> x = X0+8h+g, y = Y0+r -> two 16B f32 stores.
  float* __restrict__ orow =
      soft + (size_t)img * DIM * DIM + (size_t)(Y0 + r) * DIM + (X0 + 8 * h);
  float4 s0, s1;
  s0.x = acc[0]; s0.y = acc[1]; s0.z = acc[2]; s0.w = acc[3];
  s1.x = acc[4]; s1.y = acc[5]; s1.z = acc[6]; s1.w = acc[7];
  ((float4*)orow)[0] = s0;
  ((float4*)orow)[1] = s1;

  float lmn = acc[0], lmx = acc[0];
#pragma unroll
  for (int g = 1; g < 8; ++g) {
    lmn = fminf(lmn, acc[g]);
    lmx = fmaxf(lmx, acc[g]);
  }
#pragma unroll
  for (int m = 16; m >= 1; m >>= 1) {
    lmn = fminf(lmn, __shfl_xor(lmn, m, 32));
    lmx = fmaxf(lmx, __shfl_xor(lmx, m, 32));
  }
  if (lane == 0) {
    atomicMin(&mm[2 * img + 0], f2key(lmn));
    atomicMax(&mm[2 * img + 1], f2key(lmx));
  }
}

// ---------------------------------------------------------------------------
// Kernel 3: normalize + nan_to_num + max(soft, clip(a)); in-place on d_out.
// Exact launch: one float4 per thread, no loop, no guard.
// ---------------------------------------------------------------------------
__device__ __forceinline__ float nrm1(float s, float mn, float rden) {
  float v = (s - mn) * rden;
  if (isnan(v)) v = 0.0f;
  else if (isinf(v)) v = (v > 0.0f) ? 1.0f : 0.0f;
  return v;
}

__global__ void __launch_bounds__(256)
ha_final(const float* __restrict__ att, const int* __restrict__ mm,
         float* __restrict__ out) {
  const int v = blockIdx.x * blockDim.x + threadIdx.x;  // 4,194,304 exact
  const int img = v >> 16;                              // 65536 vec4 / image
  const float mn = key2f(mm[2 * img + 0]);
  const float mx = key2f(mm[2 * img + 1]);
  const float rden = 1.0f / fmaxf(mx - mn, EPS_);
  const float4* __restrict__ av = (const float4*)att;
  float4* __restrict__ ov       = (float4*)out;
  const float4 s4 = ov[v];
  const float4 a4 = av[v];
  float4 o;
  o.x = fmaxf(nrm1(s4.x, mn, rden), clip01(a4.x));
  o.y = fmaxf(nrm1(s4.y, mn, rden), clip01(a4.y));
  o.z = fmaxf(nrm1(s4.z, mn, rden), clip01(a4.z));
  o.w = fmaxf(nrm1(s4.w, mn, rden), clip01(a4.w));
  ov[v] = o;
}

// ---------------------------------------------------------------------------
extern "C" void kernel_launch(void* const* d_in, const int* in_sizes, int n_in,
                              void* d_out, int out_size, void* d_ws,
                              size_t ws_size, hipStream_t stream) {
  (void)in_sizes; (void)n_in; (void)out_size; (void)ws_size;
  const float* att = (const float*)d_in[0];  // [64,1,512,512] f32
  const float* gk  = (const float*)d_in[1];  // [1,1,31,31]    f32
  float* out = (float*)d_out;

  char*   ws   = (char*)d_ws;
  int*    mm   = (int*)ws;                   // 64*2 int keys (512 B)
  __bf16* wf   = (__bf16*)(ws + 512);        // 2x 32 lanes x 16 bf16 (2 KB)
  __bf16* tmpT = (__bf16*)(ws + 4096);       // 64*512 rows * 544 bf16 = 34 MB

  ha_prep <<<1,     64, 0, stream>>>(gk, wf, mm);
  ha_zhalo<<<128,  256, 0, stream>>>(tmpT);             // 32768 rows
  ha_hconv<<<8192, 256, 0, stream>>>(att, wf, tmpT);    // 1 tile / wave
  ha_vconv<<<8192, 256, 0, stream>>>(tmpT, wf, out, mm);
  ha_final<<<16384, 256, 0, stream>>>(att, mm, out);    // 1 float4 / thread
}
